// UCCModel_24309514895944
// MI455X (gfx1250) — matrix operand order repaired and checked by reading, not compiled, not run on hardware
//
#include <hip/hip_runtime.h>

// ---------------------------------------------------------------------------
// MI455X (gfx1250) implementation.
//  * per-image fused conv encoder / convT decoder, activations resident in LDS
//  * batched FC layers as f16 WMMA GEMMs (f32 accumulate), M = 8192
//    - weights pre-packed to padded f16 K-slab layout -> guard-free, vectorized
//    - 64x64 block tile, 8 waves, 2 v_wmma per wave per K-step
//  * KDE soft histogram + WMMA classifier MLP (Mpad = 64)
// ---------------------------------------------------------------------------

typedef __attribute__((ext_vector_type(16))) _Float16 v16h_t;
typedef __attribute__((ext_vector_type(8)))  float    v8f_t;
typedef __attribute__((ext_vector_type(4)))  _Float16 h4_t;

#define NIMG   8192
#define EMB    110
#define BINS   11
#define BAGS   32
#define INST   256

#define LDS_A  40   // padded f16 k-stride for sA rows
#define LDS_B  36   // padded f16 k-stride for sB rows (transposed [n][k])

// ------------------------------ WMMA GEMM ----------------------------------
// C[Mpad,Npad] = act(A[Mpad,Kpad] @ B + bias) ; A,C f32 (padded), B packed f16.
// B slab layout: elem(k,n) at ((k/32)*Npad + n)*32 + k%32 ; zero-padded.
// Grid: (Mpad/64, Npad/64); 256 threads = 8 waves as 4(M) x 2(N), each wave
// owns a 16x32 strip -> two 16x16 WMMA accumulators sharing the A fragment.
__global__ __launch_bounds__(256)
void gemm_wmma(const float* __restrict__ A, const _Float16* __restrict__ Bp,
               const float* __restrict__ bias, float* __restrict__ C,
               int lda, int ldc, int Kpad, int do_relu)
{
    __shared__ _Float16 sA[64 * LDS_A];
    __shared__ _Float16 sB[64 * LDS_B];   // transposed: [n][k]

    const int tid  = threadIdx.x;
    const int lane = tid & 31;
    const int wave = tid >> 5;
    const int wm   = wave >> 1;          // 0..3
    const int wn   = wave & 1;           // 0..1
    const int m0   = blockIdx.x * 64;
    const int n0   = blockIdx.y * 64;
    const int Npad = gridDim.y * 64;     // == ldc

    const int ar = tid >> 2;             // 0..63 : A row per staging thread
    const int ac = (tid & 3) * 8;        // 8 consecutive k
    const int bn = tid >> 2;             // 0..63 : B column per staging thread
    const int bk = (tid & 3) * 8;        // 8 consecutive k

    const int mrow = wm * 16 + (lane & 15);
    const int nl0  = wn * 32 + (lane & 15);
    const int hl   = lane >> 4;

    v8f_t acc0 = {}, acc1 = {};

    for (int k0 = 0; k0 < Kpad; k0 += 32) {
        if (k0 + 32 < Kpad)
            __builtin_prefetch(A + (long)(m0 + ar) * lda + k0 + 32, 0, 0);

        // stage A 64x32: two global_load_b128 per thread, cvt to f16
        {
            const float4* s = (const float4*)(A + (long)(m0 + ar) * lda + k0 + ac);
            float4 v0 = s[0], v1 = s[1];
            _Float16* d = &sA[ar * LDS_A + ac];
            d[0] = (_Float16)v0.x; d[1] = (_Float16)v0.y;
            d[2] = (_Float16)v0.z; d[3] = (_Float16)v0.w;
            d[4] = (_Float16)v1.x; d[5] = (_Float16)v1.y;
            d[6] = (_Float16)v1.z; d[7] = (_Float16)v1.w;
        }
        // stage B 64x32 (transposed in LDS): two 64-bit f16 loads per thread
        {
            const h4_t* s = (const h4_t*)(Bp + (((long)(k0 >> 5) * Npad + n0 + bn) << 5) + bk);
            h4_t v0 = s[0], v1 = s[1];
            h4_t* d = (h4_t*)&sB[bn * LDS_B + bk];
            d[0] = v0; d[1] = v1;
        }
        __syncthreads();

        // Fragments in contiguous k order (k = 16*hl + s), same for A and B; pairs loaded as b32.
        union { v16h_t v; unsigned int u[8]; } fa, fb0, fb1;
#pragma unroll
        for (int j = 0; j < 8; ++j) {
            int ka = hl * 16 + 2 * j;
            fa.u[j]  = *(const unsigned int*)&sA[mrow * LDS_A + ka];
            int kb = hl * 16 + 2 * j;
            fb0.u[j] = *(const unsigned int*)&sB[nl0 * LDS_B + kb];
            fb1.u[j] = *(const unsigned int*)&sB[(nl0 + 16) * LDS_B + kb];
        }
        acc0 = __builtin_amdgcn_wmma_f32_16x16x32_f16(
                   false, fa.v, false, fb0.v, (short)0, acc0, false, false);
        acc1 = __builtin_amdgcn_wmma_f32_16x16x32_f16(
                   false, fa.v, false, fb1.v, (short)0, acc1, false, false);
        __syncthreads();
    }

    const int col0 = n0 + nl0;
    const float b0 = bias[col0], b1 = bias[col0 + 16];
#pragma unroll
    for (int r = 0; r < 8; ++r) {
        int row = m0 + wm * 16 + r + hl * 8;
        float v0 = acc0[r] + b0;
        float v1 = acc1[r] + b1;
        if (do_relu) { v0 = fmaxf(v0, 0.0f); v1 = fmaxf(v1, 0.0f); }
        C[(long)row * ldc + col0]      = v0;
        C[(long)row * ldc + col0 + 16] = v1;
    }
}

// --------------------------- packing kernels -------------------------------
__global__ __launch_bounds__(256)
void pack_weight_slab(const float* __restrict__ w, _Float16* __restrict__ out,
                      int K, int N, int Kpad, int Npad)
{
    long total = (long)Kpad * Npad;
    for (long i = (long)blockIdx.x * blockDim.x + threadIdx.x; i < total;
         i += (long)gridDim.x * blockDim.x) {
        int k = (int)(i / Npad);
        int n = (int)(i - (long)k * Npad);
        float v = (k < K && n < N) ? w[(long)k * N + n] : 0.0f;
        out[(((long)(k >> 5) * Npad + n) << 5) + (k & 31)] = (_Float16)v;
    }
}

__global__ __launch_bounds__(256)
void pack_bias_pad(const float* __restrict__ b, float* __restrict__ out, int N, int Npad)
{
    int i = blockIdx.x * blockDim.x + threadIdx.x;
    if (i < Npad) out[i] = (i < N) ? b[i] : 0.0f;
}

__global__ __launch_bounds__(256)
void zero_fill(float* __restrict__ p, long n)
{
    for (long i = (long)blockIdx.x * blockDim.x + threadIdx.x; i < n;
         i += (long)gridDim.x * blockDim.x) p[i] = 0.0f;
}

__global__ __launch_bounds__(320)
void copy_logits(const float* __restrict__ s, float* __restrict__ out)
{
    int i = threadIdx.x;               // 0..319
    out[i] = s[(i / 10) * 64 + (i % 10)];
}

// --------------------------- conv helpers (LDS) ----------------------------
__device__ inline void conv3x3_relu(const float* __restrict__ in, float* __restrict__ out,
                                    const float* __restrict__ w, const float* __restrict__ bias,
                                    int Cin, int Cout, int Hin, int Win, int stride, int tid)
{
    const int Ho = Hin / stride, Wo = Win / stride;
    const int total = Cout * Ho * Wo;
    for (int idx = tid; idx < total; idx += 256) {
        int o   = idx / (Ho * Wo);
        int rem = idx - o * (Ho * Wo);
        int oy  = rem / Wo, ox = rem - oy * Wo;
        float s = bias[o];
        const int iy0 = oy * stride - 1, ix0 = ox * stride - 1;
        const float* wo = w + (long)o * Cin * 9;
        for (int i = 0; i < Cin; ++i) {
            const float* inp = in + i * Hin * Win;
            const float* wi  = wo + i * 9;
#pragma unroll
            for (int ky = 0; ky < 3; ++ky) {
                int iy = iy0 + ky;
                if (iy < 0 || iy >= Hin) continue;
#pragma unroll
                for (int kx = 0; kx < 3; ++kx) {
                    int ix = ix0 + kx;
                    if (ix < 0 || ix >= Win) continue;
                    s = fmaf(inp[iy * Win + ix], wi[ky * 3 + kx], s);
                }
            }
        }
        out[idx] = fmaxf(s, 0.0f);
    }
}

// ConvTranspose2d(k=3, pad=1, out_pad=s-1) == lhs-dilated conv, pad (1, s).
__device__ inline void convT3x3(const float* __restrict__ in, float* __restrict__ out,
                                const float* __restrict__ w, const float* __restrict__ bias,
                                int Cin, int Cout, int Hin, int Win, int s, int tid)
{
    const int Ho = Hin * s, Wo = Win * s;
    const int total = Cout * Ho * Wo;
    for (int idx = tid; idx < total; idx += 256) {
        int o   = idx / (Ho * Wo);
        int rem = idx - o * (Ho * Wo);
        int oy  = rem / Wo, ox = rem - oy * Wo;
        float acc = bias[o];
        for (int i = 0; i < Cin; ++i) {
            const float* inp = in + i * Hin * Win;
            const float* wi  = w + ((long)o * Cin + i) * 9;
#pragma unroll
            for (int ky = 0; ky < 3; ++ky) {
                int dy = oy + ky - 1;
                if (dy < 0 || (dy % s) != 0) continue;
                int iy = dy / s;
                if (iy >= Hin) continue;
#pragma unroll
                for (int kx = 0; kx < 3; ++kx) {
                    int dx = ox + kx - 1;
                    if (dx < 0 || (dx % s) != 0) continue;
                    int ix = dx / s;
                    if (ix >= Win) continue;
                    acc = fmaf(inp[iy * Win + ix], wi[ky * 3 + kx], acc);
                }
            }
        }
        out[idx] = fmaxf(acc, 0.0f);   // every decoder convT in reference has relu
    }
}

// ------------------------- encoder conv pipeline ---------------------------
__global__ __launch_bounds__(256)
void encoder_conv_kernel(const float* __restrict__ x,
                         const float* cw1, const float* cb1,
                         const float* cw2, const float* cb2,
                         const float* cw3, const float* cb3,
                         const float* cw4, const float* cb4,
                         const float* cw5, const float* cb5,
                         float* __restrict__ h0)
{
    __shared__ float bufBig[18 * 32 * 32];  // 73.7 KB
    __shared__ float bufSml[18 * 16 * 16];  // 18.4 KB  (92 KB total < 320 KB/WGP)
    const int img = blockIdx.x;
    const int tid = threadIdx.x;

    const float* xi = x + (long)img * 3 * 32 * 32;
    for (int i = tid; i < 3 * 32 * 32; i += 256) bufSml[i] = xi[i];
    __syncthreads();
    conv3x3_relu(bufSml, bufBig, cw1, cb1,  3, 18, 32, 32, 1, tid);  __syncthreads();
    conv3x3_relu(bufBig, bufSml, cw2, cb2, 18, 18, 32, 32, 2, tid);  __syncthreads();
    conv3x3_relu(bufSml, bufBig, cw3, cb3, 18, 18, 16, 16, 1, tid);  __syncthreads();
    conv3x3_relu(bufBig, bufSml, cw4, cb4, 18,  9, 16, 16, 1, tid);  __syncthreads();
    conv3x3_relu(bufSml, bufBig, cw5, cb5,  9,  9, 16, 16, 2, tid);  __syncthreads();

    float* h = h0 + (long)img * 576;
    for (int i = tid; i < 576; i += 256) h[i] = bufBig[i];
}

// ------------------------- decoder convT pipeline --------------------------
__global__ __launch_bounds__(256)
void decoder_convT_kernel(const float* __restrict__ d3,
                          const float* tw1, const float* tb1,
                          const float* tw2, const float* tb2,
                          const float* tw3, const float* tb3,
                          const float* tw4, const float* tb4,
                          const float* tw5, const float* tb5,
                          float* __restrict__ decoded)
{
    __shared__ float bufBig[18 * 32 * 32];
    __shared__ float bufSml[18 * 16 * 16];
    const int img = blockIdx.x;
    const int tid = threadIdx.x;

    for (int i = tid; i < 576; i += 256) bufBig[i] = d3[(long)img * 576 + i];
    __syncthreads();
    convT3x3(bufBig, bufSml, tw1, tb1,  9,  9,  8,  8, 2, tid);  __syncthreads(); // 9x16x16
    convT3x3(bufSml, bufBig, tw2, tb2,  9,  9, 16, 16, 1, tid);  __syncthreads(); // 9x16x16
    convT3x3(bufBig, bufSml, tw3, tb3,  9, 18, 16, 16, 1, tid);  __syncthreads(); // 18x16x16
    convT3x3(bufSml, bufBig, tw4, tb4, 18, 18, 16, 16, 2, tid);  __syncthreads(); // 18x32x32
    convT3x3(bufBig, decoded + (long)img * 3 * 32 * 32, tw5, tb5, 18, 3, 32, 32, 1, tid);
}

// ------------------------------- KDE ---------------------------------------
// emb has padded row stride 128 (cols 110..127 are exact zeros).
__global__ __launch_bounds__(128)
void kde_kernel(const float* __restrict__ emb, float* __restrict__ feat)
{
    const int b = blockIdx.x;      // bag
    const int f = threadIdx.x;     // feature
    if (f >= EMB) return;
    const float coef = -50.0f;     // -1/(2*0.1^2); norm const cancels in normalize
    float dens[BINS];
#pragma unroll
    for (int j = 0; j < BINS; ++j) dens[j] = 0.0f;
    for (int i = 0; i < INST; ++i) {
        float e = emb[((long)b * INST + i) * 128 + f];
#pragma unroll
        for (int j = 0; j < BINS; ++j) {
            float d = 0.1f * (float)j - e;
            dens[j] += __expf(coef * d * d);
        }
    }
    float tot = 0.0f;
#pragma unroll
    for (int j = 0; j < BINS; ++j) tot += dens[j];
    float inv = 1.0f / tot;
#pragma unroll
    for (int j = 0; j < BINS; ++j)
        feat[(long)b * 1216 + f * BINS + j] = dens[j] * inv;
}

// ------------------------------ launcher -----------------------------------
extern "C" void kernel_launch(void* const* d_in, const int* in_sizes, int n_in,
                              void* d_out, int out_size, void* d_ws, size_t ws_size,
                              hipStream_t stream)
{
    const float* x   = (const float*)d_in[0];
    const float* cw1 = (const float*)d_in[1];  const float* cb1 = (const float*)d_in[2];
    const float* cw2 = (const float*)d_in[3];  const float* cb2 = (const float*)d_in[4];
    const float* cw3 = (const float*)d_in[5];  const float* cb3 = (const float*)d_in[6];
    const float* cw4 = (const float*)d_in[7];  const float* cb4 = (const float*)d_in[8];
    const float* cw5 = (const float*)d_in[9];  const float* cb5 = (const float*)d_in[10];
    const float* ew1 = (const float*)d_in[11]; const float* eb1 = (const float*)d_in[12];
    const float* ew2 = (const float*)d_in[13]; const float* eb2 = (const float*)d_in[14];
    const float* ew3 = (const float*)d_in[15]; const float* eb3 = (const float*)d_in[16];
    const float* dw1 = (const float*)d_in[17]; const float* db1 = (const float*)d_in[18];
    const float* dw2 = (const float*)d_in[19]; const float* db2 = (const float*)d_in[20];
    const float* dw3 = (const float*)d_in[21]; const float* db3 = (const float*)d_in[22];
    const float* tw1 = (const float*)d_in[23]; const float* tb1 = (const float*)d_in[24];
    const float* tw2 = (const float*)d_in[25]; const float* tb2 = (const float*)d_in[26];
    const float* tw3 = (const float*)d_in[27]; const float* tb3 = (const float*)d_in[28];
    const float* tw4 = (const float*)d_in[29]; const float* tb4 = (const float*)d_in[30];
    const float* tw5 = (const float*)d_in[31]; const float* tb5 = (const float*)d_in[32];
    const float* mw1 = (const float*)d_in[33]; const float* mb1 = (const float*)d_in[34];
    const float* mw2 = (const float*)d_in[35]; const float* mb2 = (const float*)d_in[36];
    const float* mw3 = (const float*)d_in[37]; const float* mb3 = (const float*)d_in[38];

    // ---- workspace layout: f32 activations + padded biases, then f16 weights
    float* ws   = (float*)d_ws;
    float* h0   = ws;                         // 8192 x 576
    float* h1   = h0   + (long)NIMG * 576;    // 8192 x 576
    float* h2   = h1   + (long)NIMG * 576;    // 8192 x 320  (288 + zero pad)
    float* embv = h2   + (long)NIMG * 320;    // 8192 x 128  (110 + zero pad)
    float* dd1  = embv + (long)NIMG * 128;    // 8192 x 320
    float* dd2  = dd1  + (long)NIMG * 320;    // 8192 x 576
    float* dd3  = dd2  + (long)NIMG * 576;    // 8192 x 576
    float* feat = dd3  + (long)NIMG * 576;    // 64 x 1216 (zero-initialized)
    float* m1   = feat + 64L * 1216;          // 64 x 512
    float* m2   = m1   + 64L * 512;           // 64 x 256
    float* lg   = m2   + 64L * 256;           // 64 x 64
    float* pb   = lg   + 64L * 64;            // padded biases, 9 entries
    float* pb_eb1 = pb;            float* pb_eb2 = pb_eb1 + 576;
    float* pb_eb3 = pb_eb2 + 320;  float* pb_db1 = pb_eb3 + 128;
    float* pb_db2 = pb_db1 + 320;  float* pb_db3 = pb_db2 + 576;
    float* pb_mb1 = pb_db3 + 576;  float* pb_mb2 = pb_mb1 + 512;
    float* pb_mb3 = pb_mb2 + 256;
    _Float16* pw = (_Float16*)(pb_mb3 + 64);  // f16 weight slabs (16B aligned)
    _Float16* pw_ew1 = pw;                    _Float16* pw_ew2 = pw_ew1 + 576L * 576;
    _Float16* pw_ew3 = pw_ew2 + 576L * 320;   _Float16* pw_dw1 = pw_ew3 + 320L * 128;
    _Float16* pw_dw2 = pw_dw1 + 128L * 320;   _Float16* pw_dw3 = pw_dw2 + 320L * 576;
    _Float16* pw_mw1 = pw_dw3 + 576L * 576;   _Float16* pw_mw2 = pw_mw1 + 1216L * 512;
    _Float16* pw_mw3 = pw_mw2 + 512L * 256;

    float* logits  = (float*)d_out;           // 32 x 10
    float* decoded = (float*)d_out + 320;     // 8192 x 3 x 32 x 32

    // ---- 0) pack weights/biases into padded f16 slabs; zero feat buffer
    pack_weight_slab<<<256, 256, 0, stream>>>(ew1, pw_ew1,  576, 576,  576, 576);
    pack_weight_slab<<<256, 256, 0, stream>>>(ew2, pw_ew2,  576, 288,  576, 320);
    pack_weight_slab<<<64,  256, 0, stream>>>(ew3, pw_ew3,  288, 110,  320, 128);
    pack_weight_slab<<<64,  256, 0, stream>>>(dw1, pw_dw1,  110, 288,  128, 320);
    pack_weight_slab<<<256, 256, 0, stream>>>(dw2, pw_dw2,  288, 576,  320, 576);
    pack_weight_slab<<<256, 256, 0, stream>>>(dw3, pw_dw3,  576, 576,  576, 576);
    pack_weight_slab<<<512, 256, 0, stream>>>(mw1, pw_mw1, 1210, 512, 1216, 512);
    pack_weight_slab<<<128, 256, 0, stream>>>(mw2, pw_mw2,  512, 256,  512, 256);
    pack_weight_slab<<<16,  256, 0, stream>>>(mw3, pw_mw3,  256,  10,  256,  64);
    pack_bias_pad<<<3, 256, 0, stream>>>(eb1, pb_eb1, 576, 576);
    pack_bias_pad<<<2, 256, 0, stream>>>(eb2, pb_eb2, 288, 320);
    pack_bias_pad<<<1, 256, 0, stream>>>(eb3, pb_eb3, 110, 128);
    pack_bias_pad<<<2, 256, 0, stream>>>(db1, pb_db1, 288, 320);
    pack_bias_pad<<<3, 256, 0, stream>>>(db2, pb_db2, 576, 576);
    pack_bias_pad<<<3, 256, 0, stream>>>(db3, pb_db3, 576, 576);
    pack_bias_pad<<<2, 256, 0, stream>>>(mb1, pb_mb1, 512, 512);
    pack_bias_pad<<<1, 256, 0, stream>>>(mb2, pb_mb2, 256, 256);
    pack_bias_pad<<<1, 256, 0, stream>>>(mb3, pb_mb3, 10, 64);
    zero_fill<<<304, 256, 0, stream>>>(feat, 64L * 1216);

    // ---- 1) encoder convs (per-image, LDS resident)
    encoder_conv_kernel<<<NIMG, 256, 0, stream>>>(x, cw1, cb1, cw2, cb2, cw3, cb3,
                                                  cw4, cb4, cw5, cb5, h0);

    // ---- 2) FC chains as WMMA GEMMs (guard-free, padded)
    gemm_wmma<<<dim3(128, 9), 256, 0, stream>>>(h0,   pw_ew1, pb_eb1, h1,   576, 576,  576, 1);
    gemm_wmma<<<dim3(128, 5), 256, 0, stream>>>(h1,   pw_ew2, pb_eb2, h2,   576, 320,  576, 1);
    gemm_wmma<<<dim3(128, 2), 256, 0, stream>>>(h2,   pw_ew3, pb_eb3, embv, 320, 128,  320, 0);
    gemm_wmma<<<dim3(128, 5), 256, 0, stream>>>(embv, pw_dw1, pb_db1, dd1,  128, 320,  128, 0);
    gemm_wmma<<<dim3(128, 9), 256, 0, stream>>>(dd1,  pw_dw2, pb_db2, dd2,  320, 576,  320, 0);
    gemm_wmma<<<dim3(128, 9), 256, 0, stream>>>(dd2,  pw_dw3, pb_db3, dd3,  576, 576,  576, 0);

    // ---- 3) decoder transposed convs -> decoded output
    decoder_convT_kernel<<<NIMG, 256, 0, stream>>>(dd3, tw1, tb1, tw2, tb2, tw3, tb3,
                                                   tw4, tb4, tw5, tb5, decoded);

    // ---- 4) KDE soft histogram -> features (into zeroed padded buffer)
    kde_kernel<<<BAGS, 128, 0, stream>>>(embv, feat);

    // ---- 5) classifier MLP (Mpad = 64; junk pad rows never mix into real rows)
    gemm_wmma<<<dim3(1, 8), 256, 0, stream>>>(feat, pw_mw1, pb_mb1, m1, 1216, 512, 1216, 1);
    gemm_wmma<<<dim3(1, 4), 256, 0, stream>>>(m1,   pw_mw2, pb_mb2, m2,  512, 256,  512, 1);
    gemm_wmma<<<dim3(1, 1), 256, 0, stream>>>(m2,   pw_mw3, pb_mb3, lg,  256,  64,  256, 0);
    copy_logits<<<1, 320, 0, stream>>>(lg, logits);

    (void)in_sizes; (void)n_in; (void)out_size; (void)ws_size;
}
